// Conv2dfft_18047452578614
// MI455X (gfx1250) — matrix operand order repaired and checked by reading, not compiled, not run on hardware
//
#include <hip/hip_runtime.h>

typedef _Float16 v16h __attribute__((ext_vector_type(16)));
typedef _Float16 v8h  __attribute__((ext_vector_type(8)));
typedef float    v8f  __attribute__((ext_vector_type(8)));

#define N_     64
#define C_     128
#define H_     32
#define W_     32
#define HW_    1024
#define F_     128
#define OH_    30
#define OW_    30
#define K_     1152          // C*3*3
#define M_     57600         // N*OH*OW
#define KC     32            // K chunk (one WMMA depth)
#define NCHUNK 36            // K_/KC

// fast-path block tile
#define BM     128
#define BF     64

// workspace layout (bytes)
#define XHI_OFF 0UL
#define XLO_OFF 16777216UL        // N*HW*C * 2B
#define WHI_OFF 33554432UL
#define WLO_OFF 33849344UL        // + F*K * 2B
#define WS_NEED 34144256UL

// ---------------------------------------------------------------------------
// Pack x (NCHW f32) -> NHWC f16 hi/lo, via LDS tile transpose (coalesced R+W)
// ---------------------------------------------------------------------------
__global__ __launch_bounds__(256)
void pack_x_nhwc(const float* __restrict__ x,
                 _Float16* __restrict__ xhi, _Float16* __restrict__ xlo)
{
    __shared__ float tile[32][33];
    const int n   = blockIdx.z;
    const int c0  = blockIdx.y * 32;
    const int hw0 = blockIdx.x * 32;
    const int tx  = threadIdx.x & 31;
    const int ty  = threadIdx.x >> 5;   // 0..7
    #pragma unroll
    for (int i = 0; i < 4; ++i) {
        const int c = ty + i * 8;
        tile[c][tx] = x[((long)(n * C_ + c0 + c)) * HW_ + hw0 + tx];
    }
    __syncthreads();
    #pragma unroll
    for (int i = 0; i < 4; ++i) {
        const int hwl = ty + i * 8;
        const float v = tile[tx][hwl];
        const _Float16 h = (_Float16)v;
        const long o = ((long)n * HW_ + hw0 + hwl) * C_ + c0 + tx;
        xhi[o] = h;
        xlo[o] = (_Float16)(v - (float)h);
    }
}

// ---------------------------------------------------------------------------
// Pack w (F,C,3,3 f32) -> [f][k'] f16 hi/lo with k' = (r*3+s)*C + c
// ---------------------------------------------------------------------------
__global__ __launch_bounds__(256)
void pack_w_kprime(const float* __restrict__ w,
                   _Float16* __restrict__ whi, _Float16* __restrict__ wlo)
{
    const int t  = blockIdx.x * 256 + threadIdx.x;   // exactly F_*K_ threads
    const int f  = t / K_;
    const int kp = t - f * K_;
    const int rs = kp >> 7;          // k' = rs*128 + c
    const int c  = kp & 127;
    const float v = w[(long)f * K_ + c * 9 + rs];
    const _Float16 h = (_Float16)v;
    whi[t] = h;
    wlo[t] = (_Float16)(v - (float)h);
}

// ---------------------------------------------------------------------------
// Fast path: implicit-GEMM conv on pre-packed f16 hi/lo data.
// 256 threads = 8 wave32, block tile 128(M) x 64(F), double-buffered LDS.
// ---------------------------------------------------------------------------
__global__ __launch_bounds__(256)
void conv_wmma_packed(const _Float16* __restrict__ xhi, const _Float16* __restrict__ xlo,
                      const _Float16* __restrict__ whi, const _Float16* __restrict__ wlo,
                      const float* __restrict__ bias, float* __restrict__ out)
{
    __shared__ _Float16 sa_hi[2][BM][KC];
    __shared__ _Float16 sa_lo[2][BM][KC];
    __shared__ _Float16 sb_hi[2][BF][KC];
    __shared__ _Float16 sb_lo[2][BF][KC];

    const int tid    = threadIdx.x;
    const int lane   = tid & 31;
    const int waveId = tid >> 5;            // 0..7
    const int lanelo = lane & 15;
    const int lanehi = lane >> 4;
    const int waveM  = (waveId >> 1) * 32;  // 0,32,64,96
    const int waveF  = (waveId & 1) * 32;   // 0,32
    const int blockM = blockIdx.x * BM;
    const int blockF = blockIdx.y * BF;

    // A staging: 2 threads per row, 16 halves (32B) each
    const int  arow  = tid >> 1;            // 0..127
    const int  akoff = (tid & 1) * 16;
    const int  gm    = blockM + arow;
    const int  an    = gm / (OH_ * OW_);
    const int  ap    = gm - an * (OH_ * OW_);
    const int  aoh   = ap / OW_;
    const int  aow   = ap - aoh * OW_;
    const long abase = ((long)(an * H_ + aoh) * W_ + aow) * C_ + akoff;

    // B staging: 4 threads per row, 8 halves (16B) each
    const int  brow  = tid >> 2;            // 0..63
    const int  bkoff = (tid & 3) * 8;
    const long bbase = (long)(blockF + brow) * K_ + bkoff;

    // uniform A offset (halves) for chunk j: k' = rs*128 + c, rs = j/4, c0 = (j&3)*32
    auto a_off = [](int j) {
        const int rs = j >> 2;
        const int r  = rs / 3;
        const int s  = rs - 3 * r;
        return r * (W_ * C_) + s * C_ + (j & 3) * KC;
    };

    v8f acc[2][2] = {};

    // ---- prologue: stage chunk 0 ----
    v16h ra_h = *(const v16h*)(xhi + abase + a_off(0));
    v16h ra_l = *(const v16h*)(xlo + abase + a_off(0));
    v8h  rb_h = *(const v8h*)(whi + bbase);
    v8h  rb_l = *(const v8h*)(wlo + bbase);
    *(v16h*)&sa_hi[0][arow][akoff] = ra_h;
    *(v16h*)&sa_lo[0][arow][akoff] = ra_l;
    *(v8h*)&sb_hi[0][brow][bkoff]  = rb_h;
    *(v8h*)&sb_lo[0][brow][bkoff]  = rb_l;
    __syncthreads();

    for (int j = 0; j < NCHUNK; ++j) {
        const int  buf  = j & 1;
        const bool more = (j + 1) < NCHUNK;

        // issue next chunk's global loads before computing on this one
        if (more) {
            const int off = a_off(j + 1);
            ra_h = *(const v16h*)(xhi + abase + off);
            ra_l = *(const v16h*)(xlo + abase + off);
            rb_h = *(const v8h*)(whi + bbase + (j + 1) * KC);
            rb_l = *(const v8h*)(wlo + bbase + (j + 1) * KC);
        }

        // ---- fragments per CDNA5 WMMA layouts ----
        // A 16x32 f16: lanes<16 hold K {0..7,16..23}; lanes>=16 hold {8..15,24..31}
        v16h afh[2], afl[2];
        #pragma unroll
        for (int mi = 0; mi < 2; ++mi) {
            const _Float16* rph = &sa_hi[buf][waveM + mi * 16 + lanelo][0];
            const _Float16* rpl = &sa_lo[buf][waveM + mi * 16 + lanelo][0];
            const v8h p0 = *(const v8h*)(rph + lanehi * 8);
            const v8h p1 = *(const v8h*)(rph + 16 + lanehi * 8);
            const v8h q0 = *(const v8h*)(rpl + lanehi * 8);
            const v8h q1 = *(const v8h*)(rpl + 16 + lanehi * 8);
            #pragma unroll
            for (int u = 0; u < 8; ++u) {
                afh[mi][u] = p0[u]; afh[mi][u + 8] = p1[u];
                afl[mi][u] = q0[u]; afl[mi][u + 8] = q1[u];
            }
        }
        // B 32x16 f16: lane = column; lanes<16 hold K {0..7,16..23}; lanes>=16 hold {8..15,24..31}
        v16h bfh[2], bfl[2];
        #pragma unroll
        for (int fi = 0; fi < 2; ++fi) {
            const _Float16* bph = &sb_hi[buf][waveF + fi * 16 + lanelo][0];
            const _Float16* bpl = &sb_lo[buf][waveF + fi * 16 + lanelo][0];
            const v8h b0 = *(const v8h*)(bph + lanehi * 8);
            const v8h b1 = *(const v8h*)(bph + 16 + lanehi * 8);
            const v8h c0 = *(const v8h*)(bpl + lanehi * 8);
            const v8h c1 = *(const v8h*)(bpl + 16 + lanehi * 8);
            bfh[fi] = __builtin_shufflevector(b0, b1, 0,1,2,3,4,5,6,7,8,9,10,11,12,13,14,15);
            bfl[fi] = __builtin_shufflevector(c0, c1, 0,1,2,3,4,5,6,7,8,9,10,11,12,13,14,15);
        }

        // ---- hi/lo split: A*B ~= Ah*Bh + Ah*Bl + Al*Bh ----
        #pragma unroll
        for (int mi = 0; mi < 2; ++mi) {
            #pragma unroll
            for (int fi = 0; fi < 2; ++fi) {
                acc[mi][fi] = __builtin_amdgcn_wmma_f32_16x16x32_f16(
                    false, afh[mi], false, bfh[fi], (short)0, acc[mi][fi], false, false);
                acc[mi][fi] = __builtin_amdgcn_wmma_f32_16x16x32_f16(
                    false, afh[mi], false, bfl[fi], (short)0, acc[mi][fi], false, false);
                acc[mi][fi] = __builtin_amdgcn_wmma_f32_16x16x32_f16(
                    false, afl[mi], false, bfh[fi], (short)0, acc[mi][fi], false, false);
            }
        }

        // stage next chunk into the other buffer
        if (more) {
            const int nb = buf ^ 1;
            *(v16h*)&sa_hi[nb][arow][akoff] = ra_h;
            *(v16h*)&sa_lo[nb][arow][akoff] = ra_l;
            *(v8h*)&sb_hi[nb][brow][bkoff]  = rb_h;
            *(v8h*)&sb_lo[nb][brow][bkoff]  = rb_l;
        }
        __syncthreads();
    }

    // ---- epilogue: bias + scatter to NCHW output ----
    #pragma unroll
    for (int mi = 0; mi < 2; ++mi) {
        #pragma unroll
        for (int fi = 0; fi < 2; ++fi) {
            const int f  = blockF + waveF + fi * 16 + lanelo;
            const float bv = bias[f];
            #pragma unroll
            for (int v = 0; v < 8; ++v) {
                const int m  = blockM + waveM + mi * 16 + v + lanehi * 8;
                const int n  = m / (OH_ * OW_);
                const int p  = m - n * (OH_ * OW_);
                const int oh = p / OW_;
                const int ow = p - oh * OW_;
                out[(((long)n * F_ + f) * OH_ + oh) * OW_ + ow] = acc[mi][fi][v] + bv;
            }
        }
    }
}

// ---------------------------------------------------------------------------
// Fallback (no workspace): round-1 kernel, in-kernel conversion, NCHW gather.
// ---------------------------------------------------------------------------
__global__ __launch_bounds__(128)
void conv_igemm_wmma_f16split(const float* __restrict__ x,
                              const float* __restrict__ w,
                              const float* __restrict__ bias,
                              float* __restrict__ out)
{
    __shared__ _Float16 a_hi[64][KC];
    __shared__ _Float16 a_lo[64][KC];
    __shared__ _Float16 b_hi[64][KC];
    __shared__ _Float16 b_lo[64][KC];

    const int tid = threadIdx.x;
    const int lane = tid & 31, waveId = tid >> 5;
    const int lanelo = lane & 15, lanehi = lane >> 4;
    const int waveM = (waveId >> 1) * 32;
    const int waveF = (waveId & 1) * 32;
    const int blockM = blockIdx.x * 64;
    const int blockF = blockIdx.y * 64;

    const int fill_row = tid >> 1;
    const int fill_k0  = (tid & 1) * 16;

    const int  gm   = blockM + fill_row;
    const int  an   = gm / (OH_ * OW_);
    const int  ap   = gm - an * (OH_ * OW_);
    const int  aoh  = ap / OW_;
    const int  aow  = ap - aoh * OW_;
    const long abase = (long)an * C_ * (H_ * W_) + (long)aoh * W_ + aow;
    const float* __restrict__ wrow = w + (long)(blockF + fill_row) * K_;

    v8f acc[2][2] = {};

    for (int kc = 0; kc < K_; kc += KC) {
        {
            _Float16 hbuf[16], lbuf[16];
            #pragma unroll
            for (int i = 0; i < 16; ++i) {
                const int k  = kc + fill_k0 + i;
                const int c  = k / 9;
                const int rs = k - c * 9;
                const int r  = rs / 3;
                const int s  = rs - r * 3;
                const float v = x[abase + (long)c * (H_ * W_) + r * W_ + s];
                const _Float16 h = (_Float16)v;
                hbuf[i] = h;
                lbuf[i] = (_Float16)(v - (float)h);
            }
            *(v8h*)&a_hi[fill_row][fill_k0]     = *(v8h*)&hbuf[0];
            *(v8h*)&a_hi[fill_row][fill_k0 + 8] = *(v8h*)&hbuf[8];
            *(v8h*)&a_lo[fill_row][fill_k0]     = *(v8h*)&lbuf[0];
            *(v8h*)&a_lo[fill_row][fill_k0 + 8] = *(v8h*)&lbuf[8];
        }
        {
            _Float16 hbuf[16], lbuf[16];
            #pragma unroll
            for (int i = 0; i < 16; i += 4) {
                const float4 v4 = *(const float4*)(wrow + kc + fill_k0 + i);
                const float vv[4] = {v4.x, v4.y, v4.z, v4.w};
                #pragma unroll
                for (int u = 0; u < 4; ++u) {
                    const _Float16 h = (_Float16)vv[u];
                    hbuf[i + u] = h;
                    lbuf[i + u] = (_Float16)(vv[u] - (float)h);
                }
            }
            *(v8h*)&b_hi[fill_row][fill_k0]     = *(v8h*)&hbuf[0];
            *(v8h*)&b_hi[fill_row][fill_k0 + 8] = *(v8h*)&hbuf[8];
            *(v8h*)&b_lo[fill_row][fill_k0]     = *(v8h*)&lbuf[0];
            *(v8h*)&b_lo[fill_row][fill_k0 + 8] = *(v8h*)&lbuf[8];
        }
        __syncthreads();

        v16h afh[2], afl[2];
        #pragma unroll
        for (int mi = 0; mi < 2; ++mi) {
            const _Float16* rph = &a_hi[waveM + mi * 16 + lanelo][0];
            const _Float16* rpl = &a_lo[waveM + mi * 16 + lanelo][0];
            const v8h p0 = *(const v8h*)(rph + lanehi * 8);
            const v8h p1 = *(const v8h*)(rph + 16 + lanehi * 8);
            const v8h q0 = *(const v8h*)(rpl + lanehi * 8);
            const v8h q1 = *(const v8h*)(rpl + 16 + lanehi * 8);
            #pragma unroll
            for (int u = 0; u < 8; ++u) {
                afh[mi][u] = p0[u]; afh[mi][u + 8] = p1[u];
                afl[mi][u] = q0[u]; afl[mi][u + 8] = q1[u];
            }
        }
        v16h bfh[2], bfl[2];
        #pragma unroll
        for (int fi = 0; fi < 2; ++fi) {
            const _Float16* bph = &b_hi[waveF + fi * 16 + lanelo][0];
            const _Float16* bpl = &b_lo[waveF + fi * 16 + lanelo][0];
            const v8h b0 = *(const v8h*)(bph + lanehi * 8);
            const v8h b1 = *(const v8h*)(bph + 16 + lanehi * 8);
            const v8h c0 = *(const v8h*)(bpl + lanehi * 8);
            const v8h c1 = *(const v8h*)(bpl + 16 + lanehi * 8);
            bfh[fi] = __builtin_shufflevector(b0, b1, 0,1,2,3,4,5,6,7,8,9,10,11,12,13,14,15);
            bfl[fi] = __builtin_shufflevector(c0, c1, 0,1,2,3,4,5,6,7,8,9,10,11,12,13,14,15);
        }

        #pragma unroll
        for (int mi = 0; mi < 2; ++mi) {
            #pragma unroll
            for (int fi = 0; fi < 2; ++fi) {
                acc[mi][fi] = __builtin_amdgcn_wmma_f32_16x16x32_f16(
                    false, afh[mi], false, bfh[fi], (short)0, acc[mi][fi], false, false);
                acc[mi][fi] = __builtin_amdgcn_wmma_f32_16x16x32_f16(
                    false, afh[mi], false, bfl[fi], (short)0, acc[mi][fi], false, false);
                acc[mi][fi] = __builtin_amdgcn_wmma_f32_16x16x32_f16(
                    false, afl[mi], false, bfh[fi], (short)0, acc[mi][fi], false, false);
            }
        }
        __syncthreads();
    }

    #pragma unroll
    for (int mi = 0; mi < 2; ++mi) {
        #pragma unroll
        for (int fi = 0; fi < 2; ++fi) {
            const int f  = blockF + waveF + fi * 16 + lanelo;
            const float bv = bias[f];
            #pragma unroll
            for (int v = 0; v < 8; ++v) {
                const int m  = blockM + waveM + mi * 16 + v + lanehi * 8;
                const int n  = m / (OH_ * OW_);
                const int p  = m - n * (OH_ * OW_);
                const int oh = p / OW_;
                const int ow = p - oh * OW_;
                out[(((long)n * F_ + f) * OH_ + oh) * OW_ + ow] = acc[mi][fi][v] + bv;
            }
        }
    }
}

extern "C" void kernel_launch(void* const* d_in, const int* in_sizes, int n_in,
                              void* d_out, int out_size, void* d_ws, size_t ws_size,
                              hipStream_t stream) {
    (void)in_sizes; (void)n_in; (void)out_size;
    const float* x    = (const float*)d_in[0];
    const float* w    = (const float*)d_in[1];
    const float* bias = (const float*)d_in[2];
    float* out = (float*)d_out;

    if (d_ws != nullptr && ws_size >= WS_NEED) {
        _Float16* xhi = (_Float16*)((char*)d_ws + XHI_OFF);
        _Float16* xlo = (_Float16*)((char*)d_ws + XLO_OFF);
        _Float16* whi = (_Float16*)((char*)d_ws + WHI_OFF);
        _Float16* wlo = (_Float16*)((char*)d_ws + WLO_OFF);
        pack_x_nhwc<<<dim3(HW_ / 32, C_ / 32, N_), 256, 0, stream>>>(x, xhi, xlo);
        pack_w_kprime<<<(F_ * K_) / 256, 256, 0, stream>>>(w, whi, wlo);
        conv_wmma_packed<<<dim3(M_ / BM, F_ / BF), 256, 0, stream>>>(xhi, xlo, whi, wlo, bias, out);
    } else {
        conv_igemm_wmma_f16split<<<dim3(M_ / 64, F_ / 64), 128, 0, stream>>>(x, w, bias, out);
    }
}
